// Model_24842090840520
// MI455X (gfx1250) — matrix operand order repaired and checked by reading, not compiled, not run on hardware
//
#include <hip/hip_runtime.h>
#include <math.h>

// ---------------------------------------------------------------------------
// Model dims
// ---------------------------------------------------------------------------
constexpr int C_DIM = 1024;
constexpr int T_DIM = 128;
constexpr int B_DIM = 64;
constexpr int NEV   = 16;
constexpr int NPAR  = 57;
constexpr float NEG_SLOPE = 0.2f;

// ---------------------------------------------------------------------------
// WMMA types / helpers (gfx1250, wave32)
// ---------------------------------------------------------------------------
typedef __attribute__((ext_vector_type(16))) __bf16 v16bf;
typedef __attribute__((ext_vector_type(8)))  float  v8f;
typedef unsigned int v4u_ __attribute__((ext_vector_type(4)));
typedef int          v8i_ __attribute__((ext_vector_type(8)));
typedef int          v4i_ __attribute__((ext_vector_type(4)));

#if defined(__has_builtin)
#  if __has_builtin(__builtin_amdgcn_tensor_load_to_lds)
#    define USE_TDM 1
#  endif
#endif
#ifndef USE_TDM
#  define USE_TDM 0
#endif

struct alignas(16) U4  { unsigned x, y, z, w; };
struct alignas(16) F4  { float    x, y, z, w; };
struct alignas(16) Raw32 { U4 lo, hi; };   // 32 bytes == v16bf

// A fragment: row-major MxK weights W[row][k], ld = row stride (elements).
// Lane (m = lane&15, g = lane>>4). ISA A-layout: K(e,g) = (e<8?0:16) + 8g + (e&7)
// -> two contiguous 16B chunks per lane.
__device__ __forceinline__ v16bf load_fragA(const unsigned short* __restrict__ base,
                                            int row, int ld, int k0, int g) {
    const unsigned short* p = base + (size_t)row * ld + k0 + g * 8;
    Raw32 r;
    r.lo = *reinterpret_cast<const U4*>(p);
    r.hi = *reinterpret_cast<const U4*>(p + 16);
    return __builtin_bit_cast(v16bf, r);
}

// B fragment from LDS tile [16 rows][C], row = n (col in tile).
// B-layout matching A: K(e,g) = (e<8?0:16) + 8g + (e&7) -> two 16B chunks per lane.
__device__ __forceinline__ v16bf load_fragB_lds(const unsigned short* p_row, int k0, int g) {
    const unsigned short* p = p_row + k0 + g * 8;
    Raw32 r;
    r.lo = *reinterpret_cast<const U4*>(p);
    r.hi = *reinterpret_cast<const U4*>(p + 16);
    return __builtin_bit_cast(v16bf, r);
}

__device__ __forceinline__ unsigned short f2bf(float f) {
    unsigned u = __builtin_bit_cast(unsigned, f);
    u += 0x7FFFu + ((u >> 16) & 1u);   // round-to-nearest-even
    return (unsigned short)(u >> 16);
}

__device__ __forceinline__ void store8bf(unsigned short* p, const float* y) {
    U4 pk;
    pk.x = (unsigned)f2bf(y[0]) | ((unsigned)f2bf(y[1]) << 16);
    pk.y = (unsigned)f2bf(y[2]) | ((unsigned)f2bf(y[3]) << 16);
    pk.z = (unsigned)f2bf(y[4]) | ((unsigned)f2bf(y[5]) << 16);
    pk.w = (unsigned)f2bf(y[6]) | ((unsigned)f2bf(y[7]) << 16);
    *reinterpret_cast<U4*>(p) = pk;
}

#if USE_TDM
// Issue a TDM 2D tile load: 16 rows x C_DIM bf16 from rows [row0, row0+16) of a
// [nrows][C_DIM] bf16 tensor into LDS at lds_off. Rows beyond nrows zero-fill
// (tensor_dim1 counts rows remaining from the tile start).
__device__ __forceinline__ void tdm_load_rows_to_lds(unsigned lds_off,
                                                     const unsigned short* gbase,
                                                     int row0, int nrows) {
    int start  = row0 < nrows ? row0 : nrows;
    int remain = nrows - start;                           // >= 0
    unsigned long long ga =
        (unsigned long long)(const void*)(gbase + (size_t)start * C_DIM);
    v4u_ g0;
    g0[0] = 1u;                                           // count=1, user mode
    g0[1] = lds_off;                                      // D#.lds_addr
    g0[2] = (unsigned)(ga & 0xffffffffull);               // global_addr[31:0]
    g0[3] = (unsigned)((ga >> 32) & 0x01ffffffull)        // global_addr[56:32]
          | (2u << 30);                                   // type = 2 ("image")
    v8i_ g1;
    g1[0] = (int)(1u << 16);                              // data_size = 1 -> 2 bytes
    g1[1] = (int)(((unsigned)C_DIM & 0xffffu) << 16);     // tensor_dim0[15:0]
    g1[2] = (int)((((unsigned)C_DIM >> 16) & 0xffffu)     // tensor_dim0[31:16]
          | (((unsigned)remain & 0xffffu) << 16));        // tensor_dim1[15:0]
    g1[3] = (int)((((unsigned)remain >> 16) & 0xffffu)    // tensor_dim1[31:16]
          | (((unsigned)C_DIM & 0xffffu) << 16));         // tile_dim0 = 1024
    g1[4] = 16;                                           // tile_dim1 = 16, tile_dim2 = 0
    g1[5] = C_DIM;                                        // tensor_dim0_stride[31:0]
    g1[6] = 0;
    g1[7] = 0;
    v4i_ z4 = {0, 0, 0, 0};
#if __clang_major__ >= 23
    v8i_ z8 = {0, 0, 0, 0, 0, 0, 0, 0};
    __builtin_amdgcn_tensor_load_to_lds(g0, g1, z4, z4, z8, 0);
#else
    __builtin_amdgcn_tensor_load_to_lds(g0, g1, z4, z4, 0);
#endif
}
#endif

// ---------------------------------------------------------------------------
// Fused WMMA GEMM.  MODE 0: spec (bias+pos, write h f32+bf16)
//                   MODE 1: conv (2 mats, leaky+residual+L2norm+acc)
//                   MODE 2: vec  (bias, write f32 only)
// grid = (T/16, B), 256 threads (8 wave32). Workgroup owns all 1024 output
// channels for a 16-timestep tile. B operand staged in LDS via the Tensor
// Data Mover (zero-fill past T = anti-causal padding). A fragments are
// double-buffered in registers: a 16-load clause for the next 8-WMMA group
// issues while the current group executes on the XDL pipe.
// ---------------------------------------------------------------------------
template <int MODE>
__global__ __launch_bounds__(256) void gemm_kernel(
    const unsigned short* __restrict__ Wa,    // [C][C] bf16 (W0)
    const unsigned short* __restrict__ Wb,    // [C][C] bf16 (W1, conv only)
    const unsigned short* __restrict__ Bmat,  // [B][T][C] bf16 activations
    const float* __restrict__ bias,           // [C]
    const float* __restrict__ pos,            // [T][C] (spec only)
    const float* __restrict__ hin_f,          // [B][T][C] residual in (conv)
    float* __restrict__ out_f,                // [B][T][C]
    unsigned short* __restrict__ out_bf,      // [B][T][C] (spec/conv)
    float* __restrict__ acc,                  // [B][T][C] (conv)
    int d)                                    // dilation
{
    extern __shared__ unsigned short smem[];  // [2][16][C_DIM] bf16 = 64KB
    unsigned short* btile0 = smem;
    unsigned short* btile1 = smem + 16 * C_DIM;

    const int tid  = threadIdx.x;
    const int wv   = tid >> 5;
    const int lane = tid & 31;
    const int n    = lane & 15;
    const int g    = lane >> 4;
    const int t0   = blockIdx.x * 16;
    const int b    = blockIdx.y;
    const int t    = t0 + n;
    const int m_base = wv * 128;              // 8 m-tiles of 16 per wave

    const unsigned short* Bp = Bmat + (size_t)b * T_DIM * C_DIM;

    // ---- stage B operand tiles into LDS ----
#if USE_TDM
    if (wv == 0) {
        unsigned lds0 = (unsigned)(unsigned long long)(const void*)btile0;
        tdm_load_rows_to_lds(lds0, Bp, t0, T_DIM);
        if (MODE == 1) {
            unsigned lds1 = (unsigned)(unsigned long long)(const void*)btile1;
            tdm_load_rows_to_lds(lds1, Bp, t0 + d, T_DIM);
        }
        __builtin_amdgcn_s_wait_tensorcnt(0);
    }
    __syncthreads();
#else
    {
        const int ntile = (MODE == 1) ? 2 : 1;
        for (int ti = 0; ti < ntile; ti++) {
            int row0 = t0 + (ti ? d : 0);
            unsigned short* dst = ti ? btile1 : btile0;
            for (int e = tid; e < 16 * (C_DIM / 8); e += 256) {
                int r  = e / (C_DIM / 8);
                int c8 = (e % (C_DIM / 8)) * 8;
                U4 v{};
                int row = row0 + r;
                if (row < T_DIM)
                    v = *reinterpret_cast<const U4*>(Bp + (size_t)row * C_DIM + c8);
                *reinterpret_cast<U4*>(dst + (size_t)r * C_DIM + c8) = v;
            }
        }
    }
    __syncthreads();
#endif

    v8f accv[8];
#pragma unroll
    for (int j = 0; j < 8; j++)
#pragma unroll
        for (int r = 0; r < 8; r++) accv[j][r] = 0.0f;

    const unsigned short* brow0 = btile0 + (size_t)n * C_DIM;
    const unsigned short* brow1 = btile1 + (size_t)n * C_DIM;
    const int arow = m_base + n;              // A lane row base (m = lane&15)

    v16bf aCur[8], aNxt[8];
#pragma unroll
    for (int j = 0; j < 8; j++)
        aCur[j] = load_fragA(Wa, arow + j * 16, C_DIM, 0, g);

    if (MODE == 1) {
        // stream: W0@k0 -> W1@k0 -> W0@k0+32 -> ...; ping-pong aCur/aNxt
        for (int k0 = 0; k0 < C_DIM; k0 += 32) {
            v16bf b0 = load_fragB_lds(brow0, k0, g);
#pragma unroll
            for (int j = 0; j < 8; j++)
                aNxt[j] = load_fragA(Wb, arow + j * 16, C_DIM, k0, g);
#pragma unroll
            for (int j = 0; j < 8; j++)
                accv[j] = __builtin_amdgcn_wmma_f32_16x16x32_bf16(
                    false, aCur[j], false, b0, (short)0, accv[j], false, false);

            v16bf b1 = load_fragB_lds(brow1, k0, g);
            if (k0 + 32 < C_DIM) {
                __builtin_prefetch(Wa + (size_t)arow * C_DIM + k0 + 64, 0, 1);
#pragma unroll
                for (int j = 0; j < 8; j++)
                    aCur[j] = load_fragA(Wa, arow + j * 16, C_DIM, k0 + 32, g);
            }
#pragma unroll
            for (int j = 0; j < 8; j++)
                accv[j] = __builtin_amdgcn_wmma_f32_16x16x32_bf16(
                    false, aNxt[j], false, b1, (short)0, accv[j], false, false);
        }
    } else {
        // single matrix: unroll K by 64, alternate aCur/aNxt
        for (int k0 = 0; k0 < C_DIM; k0 += 64) {
            v16bf b0 = load_fragB_lds(brow0, k0, g);
#pragma unroll
            for (int j = 0; j < 8; j++)
                aNxt[j] = load_fragA(Wa, arow + j * 16, C_DIM, k0 + 32, g);
#pragma unroll
            for (int j = 0; j < 8; j++)
                accv[j] = __builtin_amdgcn_wmma_f32_16x16x32_bf16(
                    false, aCur[j], false, b0, (short)0, accv[j], false, false);

            v16bf b1 = load_fragB_lds(brow0, k0 + 32, g);
            if (k0 + 64 < C_DIM) {
                __builtin_prefetch(Wa + (size_t)arow * C_DIM + k0 + 96, 0, 1);
#pragma unroll
                for (int j = 0; j < 8; j++)
                    aCur[j] = load_fragA(Wa, arow + j * 16, C_DIM, k0 + 64, g);
            }
#pragma unroll
            for (int j = 0; j < 8; j++)
                accv[j] = __builtin_amdgcn_wmma_f32_16x16x32_bf16(
                    false, aNxt[j], false, b1, (short)0, accv[j], false, false);
        }
    }

    const size_t obase = ((size_t)b * T_DIM + t) * C_DIM;

    if (MODE == 1) {
        // leaky_relu(conv + bias) + skip, then unit-norm over channels, acc +=
        float ssp = 0.0f;
#pragma unroll
        for (int j = 0; j < 8; j++) {
            int o0 = m_base + j * 16 + g * 8;
            const float* skip = hin_f + obase + o0;
            F4 s0 = *reinterpret_cast<const F4*>(skip);
            F4 s1 = *reinterpret_cast<const F4*>(skip + 4);
            float sk[8] = {s0.x, s0.y, s0.z, s0.w, s1.x, s1.y, s1.z, s1.w};
#pragma unroll
            for (int r = 0; r < 8; r++) {
                float gv = accv[j][r] + bias[o0 + r];
                gv = (gv >= 0.0f) ? gv : NEG_SLOPE * gv;
                float yv = gv + sk[r];
                accv[j][r] = yv;
                ssp += yv * yv;
            }
        }
        // reuse dead B-tile LDS for the per-timestep norm reduction
        float* ssl = reinterpret_cast<float*>(smem);      // [16][17]
        __syncthreads();                                  // btile reads complete
        ssl[n * 17 + wv * 2 + g] = ssp;
        __syncthreads();
        float tot = 0.0f;
#pragma unroll
        for (int s = 0; s < 16; s++) tot += ssl[n * 17 + s];
        float inv = 1.0f / (sqrtf(tot) + 1e-8f);

#pragma unroll
        for (int j = 0; j < 8; j++) {
            int o0 = m_base + j * 16 + g * 8;
            size_t base = obase + o0;
            float yv[8];
#pragma unroll
            for (int r = 0; r < 8; r++) yv[r] = accv[j][r] * inv;

            *reinterpret_cast<F4*>(out_f + base)     = F4{yv[0], yv[1], yv[2], yv[3]};
            *reinterpret_cast<F4*>(out_f + base + 4) = F4{yv[4], yv[5], yv[6], yv[7]};
            store8bf(out_bf + base, yv);

            F4 a0 = *reinterpret_cast<const F4*>(acc + base);
            F4 a1 = *reinterpret_cast<const F4*>(acc + base + 4);
            a0.x += yv[0]; a0.y += yv[1]; a0.z += yv[2]; a0.w += yv[3];
            a1.x += yv[4]; a1.y += yv[5]; a1.z += yv[6]; a1.w += yv[7];
            *reinterpret_cast<F4*>(acc + base)     = a0;
            *reinterpret_cast<F4*>(acc + base + 4) = a1;
        }
    } else {
#pragma unroll
        for (int j = 0; j < 8; j++) {
            int o0 = m_base + j * 16 + g * 8;
            size_t base = obase + o0;
            float yv[8];
            const float* pp = (MODE == 0) ? (pos + (size_t)t * C_DIM + o0) : nullptr;
#pragma unroll
            for (int r = 0; r < 8; r++) {
                float v = accv[j][r] + bias[o0 + r];
                if (MODE == 0) v += pp[r];
                yv[r] = v;
            }
            *reinterpret_cast<F4*>(out_f + base)     = F4{yv[0], yv[1], yv[2], yv[3]};
            *reinterpret_cast<F4*>(out_f + base + 4) = F4{yv[4], yv[5], yv[6], yv[7]};
            if (MODE == 0) store8bf(out_bf + base, yv);
        }
    }
}

// ---------------------------------------------------------------------------
// Prep / small kernels
// ---------------------------------------------------------------------------
__global__ void bf16_copy_kernel(const float* __restrict__ src,
                                 unsigned short* __restrict__ dst, int nElem) {
    int i = blockIdx.x * 256 + threadIdx.x;
    if (i < nElem) dst[i] = f2bf(src[i]);
}

__global__ void enc_convert_kernel(const float* __restrict__ src,
                                   unsigned short* __restrict__ dst) {
    int idx = blockIdx.x * 256 + threadIdx.x;   // 8*1024*1024*2
    if (idx >= 8 * C_DIM * C_DIM * 2) return;
    int k = idx & 1;
    int c = (idx >> 1) & (C_DIM - 1);
    int o = (idx >> 11) & (C_DIM - 1);
    int i = idx >> 21;
    dst[(size_t)(i * 2 + k) * C_DIM * C_DIM + (size_t)o * C_DIM + c] = f2bf(src[idx]);
}

__global__ void x_transpose_kernel(const float* __restrict__ x,
                                   unsigned short* __restrict__ dst) {
    int idx = blockIdx.x * 256 + threadIdx.x;   // B*C*T
    if (idx >= B_DIM * C_DIM * T_DIM) return;
    int t = idx & (T_DIM - 1);
    int c = (idx >> 7) & (C_DIM - 1);
    int b = idx >> 17;
    dst[((size_t)b * T_DIM + t) * C_DIM + c] = f2bf(x[idx]);
}

__global__ void pos_kernel(const float* __restrict__ w_pos,
                           const float* __restrict__ b_pos,
                           float* __restrict__ pos_out) {  // [T][C]
    int idx = blockIdx.x * 256 + threadIdx.x;
    if (idx >= T_DIM * C_DIM) return;
    int t = idx / C_DIM;
    int o = idx % C_DIM;
    float p = -1.0f + 2.0f * (float)t / (float)(T_DIM - 1);
    const float* w = w_pos + (size_t)o * 33;
    float s = w[0] * p;
    float scale = 1.0f;
#pragma unroll
    for (int i = 0; i < 16; i++) {
        float a = scale * p;
        s += w[1 + 2 * i] * sinf(a) + w[2 + 2 * i] * cosf(a);
        scale *= 2.0f;
    }
    pos_out[idx] = s + b_pos[o];
}

__global__ void zero_kernel(float* __restrict__ p, int nElem) {
    int i = blockIdx.x * 256 + threadIdx.x;
    if (i < nElem) p[i] = 0.0f;
}

// switch = w_sw . acc + b_sw ; attn = |switch| ; serial stable top-16 per batch
__global__ void switch_topk_kernel(const float* __restrict__ acc,  // [B][T][C]
                                   const float* __restrict__ w_sw,
                                   const float* __restrict__ b_sw,
                                   int* __restrict__ idx_out,
                                   float* __restrict__ val_out) {
    __shared__ float av[T_DIM];
    int b = blockIdx.x;
    int t = threadIdx.x;
    const float* col = acc + ((size_t)b * T_DIM + t) * C_DIM;
    float s = 0.0f;
    for (int c = 0; c < C_DIM; c++) s += w_sw[c] * col[c];
    s += b_sw[0];
    av[t] = fabsf(s);
    __syncthreads();
    if (t == 0) {
        for (int e = 0; e < NEV; e++) {
            float best = -1.0f; int bi = 0;
            for (int i = 0; i < T_DIM; i++)
                if (av[i] > best) { best = av[i]; bi = i; }   // ties -> lowest idx
            idx_out[b * NEV + e] = bi;
            val_out[b * NEV + e] = best;
            av[bi] = -2.0f;
        }
    }
}

// out[b,e,p] = val[b,e] * (vecs[b,:,idx] . w_par[p,:]) + b_par[p]
__global__ void params_kernel(const float* __restrict__ vecs,  // [B][T][C]
                              const int* __restrict__ idx,
                              const float* __restrict__ val,
                              const float* __restrict__ w_par,
                              const float* __restrict__ b_par,
                              float* __restrict__ out) {
    int b = blockIdx.x, e = blockIdx.y, p = threadIdx.x;
    if (p >= NPAR) return;
    int t = idx[b * NEV + e];
    const float* v = vecs + ((size_t)b * T_DIM + t) * C_DIM;
    const float* w = w_par + (size_t)p * C_DIM;
    float s = 0.0f;
    for (int c = 0; c < C_DIM; c++) s += v[c] * w[c];
    out[((size_t)b * NEV + e) * NPAR + p] = val[b * NEV + e] * s + b_par[p];
}

// ---------------------------------------------------------------------------
// Host launch
// ---------------------------------------------------------------------------
extern "C" void kernel_launch(void* const* d_in, const int* in_sizes, int n_in,
                              void* d_out, int out_size, void* d_ws, size_t ws_size,
                              hipStream_t stream) {
    (void)in_sizes; (void)n_in; (void)out_size; (void)ws_size;

    const float* x      = (const float*)d_in[0];
    const float* w_spec = (const float*)d_in[1];
    const float* b_spec = (const float*)d_in[2];
    const float* w_pos  = (const float*)d_in[3];
    const float* b_pos  = (const float*)d_in[4];
    const float* enc_w  = (const float*)d_in[5];
    const float* enc_b  = (const float*)d_in[6];
    const float* w_vec  = (const float*)d_in[7];
    const float* b_vec  = (const float*)d_in[8];
    const float* w_sw   = (const float*)d_in[9];
    const float* b_sw   = (const float*)d_in[10];
    const float* w_par  = (const float*)d_in[11];
    const float* b_par  = (const float*)d_in[12];
    float* out = (float*)d_out;

    char* ws = (char*)d_ws;
    size_t off = 0;
    auto alloc = [&](size_t bytes) {
        size_t o = off;
        off += (bytes + 255) & ~(size_t)255;
        return o;
    };
    const size_t NBT  = (size_t)B_DIM * T_DIM * C_DIM;          // 8,388,608 elems
    const size_t o_xbf   = alloc(NBT * 2);                      // x bf16 [b][t][c]; reused as acc_bf
    const size_t o_wspec = alloc((size_t)C_DIM * C_DIM * 2);
    const size_t o_wvec  = alloc((size_t)C_DIM * C_DIM * 2);
    const size_t o_wenc  = alloc((size_t)16 * C_DIM * C_DIM * 2);
    const size_t o_pos   = alloc((size_t)T_DIM * C_DIM * 4);
    const size_t o_h0f   = alloc(NBT * 4);
    const size_t o_h1f   = alloc(NBT * 4);                      // reused as vecs
    const size_t o_h0b   = alloc(NBT * 2);
    const size_t o_h1b   = alloc(NBT * 2);
    const size_t o_acc   = alloc(NBT * 4);
    const size_t o_idx   = alloc((size_t)B_DIM * NEV * 4);
    const size_t o_val   = alloc((size_t)B_DIM * NEV * 4);

    unsigned short* xbf    = (unsigned short*)(ws + o_xbf);
    unsigned short* wspecb = (unsigned short*)(ws + o_wspec);
    unsigned short* wvecb  = (unsigned short*)(ws + o_wvec);
    unsigned short* wencb  = (unsigned short*)(ws + o_wenc);
    float*          posf   = (float*)(ws + o_pos);
    float*          hf[2]  = {(float*)(ws + o_h0f), (float*)(ws + o_h1f)};
    unsigned short* hb[2]  = {(unsigned short*)(ws + o_h0b), (unsigned short*)(ws + o_h1b)};
    float*          accf   = (float*)(ws + o_acc);
    unsigned short* accbf  = xbf;                 // x dead after spec gemm
    float*          vecs   = hf[1];               // slab 1 dead after layer 8
    int*            idxb   = (int*)(ws + o_idx);
    float*          valb   = (float*)(ws + o_val);

    auto blocks = [](size_t n) { return (unsigned)((n + 255) / 256); };
    const unsigned smemBytes = 2u * 16u * C_DIM * 2u;           // 64KB dynamic LDS

    // --- prep ---
    bf16_copy_kernel<<<blocks(C_DIM * C_DIM), 256, 0, stream>>>(w_spec, wspecb, C_DIM * C_DIM);
    bf16_copy_kernel<<<blocks(C_DIM * C_DIM), 256, 0, stream>>>(w_vec, wvecb, C_DIM * C_DIM);
    enc_convert_kernel<<<blocks((size_t)8 * C_DIM * C_DIM * 2), 256, 0, stream>>>(enc_w, wencb);
    x_transpose_kernel<<<blocks(NBT), 256, 0, stream>>>(x, xbf);
    pos_kernel<<<blocks((size_t)T_DIM * C_DIM), 256, 0, stream>>>(w_pos, b_pos, posf);
    zero_kernel<<<blocks(NBT), 256, 0, stream>>>(accf, (int)NBT);

    dim3 gg(T_DIM / 16, B_DIM);

    // --- h = w_spec.x + b_spec + pos ---
    gemm_kernel<0><<<gg, 256, smemBytes, stream>>>(wspecb, nullptr, xbf, b_spec, posf,
                                                   nullptr, hf[0], hb[0], nullptr, 0);

    // --- anti-causal dilated stack ---
    const int dil[8] = {1, 2, 4, 8, 16, 32, 64, 1};
    int cur = 0;
    for (int i = 0; i < 8; i++) {
        int nxt = cur ^ 1;
        const unsigned short* W0 = wencb + (size_t)(i * 2)     * C_DIM * C_DIM;
        const unsigned short* W1 = wencb + (size_t)(i * 2 + 1) * C_DIM * C_DIM;
        gemm_kernel<1><<<gg, 256, smemBytes, stream>>>(W0, W1, hb[cur],
                                                       enc_b + (size_t)i * C_DIM,
                                                       nullptr, hf[cur], hf[nxt], hb[nxt],
                                                       accf, dil[i]);
        cur = nxt;
    }

    // --- vecs = w_vec . encoded + b_vec ---
    bf16_copy_kernel<<<blocks(NBT), 256, 0, stream>>>(accf, accbf, (int)NBT);
    gemm_kernel<2><<<gg, 256, smemBytes, stream>>>(wvecb, nullptr, accbf, b_vec, nullptr,
                                                   nullptr, vecs, nullptr, nullptr, 0);

    // --- switch -> |.| -> top-16 ---
    switch_topk_kernel<<<B_DIM, T_DIM, 0, stream>>>(accf, w_sw, b_sw, idxb, valb);

    // --- events -> params ---
    params_kernel<<<dim3(B_DIM, NEV), 64, 0, stream>>>(vecs, idxb, valb, w_par, b_par, out);
}
